// MetaLayer_2551210573871
// MI455X (gfx1250) — matrix operand order repaired and checked by reading, not compiled, not run on hardware
//
#include <hip/hip_runtime.h>
#include <hip/hip_bf16.h>

// ---- problem constants (match reference) ----
#define S_DIM   128
#define IN_DIM  256
#define OUT_DIM 256
#define BATCH   4096
#define K1      (S_DIM * IN_DIM)      // 32768 : T1 region of K
#define KTOT    (K1 + IN_DIM)         // 33024 : + M1 bias region
#define NCHUNKS (KTOT / 32)           // 1032 K-chunks of 32
#define KSPLIT  4
#define CPS     (NCHUNKS / KSPLIT)    // 258 chunks per K-slice

// ---- tiling ----
#define MTILE   64                    // rows per workgroup
#define MBLOCKS (BATCH / MTILE)       // 64

// ---- LDS pitches (halfs), padded to stagger banks ----
#define FEAT_PITCH 272                // 256 cols + pad
#define SIG_PITCH  136                // 129 cols (s=128 is the "ones" column) + pad

// pre-swizzled B: [NCHUNKS][16 ntiles][32 lanes][16 halfs]
#define W_HALFS ((size_t)NCHUNKS * 16 * 32 * 16)
#define W_BYTES (W_HALFS * 2)         // 16,908,288 B

typedef __attribute__((ext_vector_type(16))) _Float16 v16h;
typedef __attribute__((ext_vector_type(8)))  _Float16 v8h;
typedef __attribute__((ext_vector_type(4)))  _Float16 v4h;
typedef __attribute__((ext_vector_type(8)))  float    v8f;

// --------------------------------------------------------------------------
// One-shot: convert T1||M1 (f32) into f16 in exact WMMA B-fragment order.
// Work item = (chunk c, ntile) handled by one wave; lane writes its 32B frag.
// --------------------------------------------------------------------------
__global__ void __launch_bounds__(256)
swizzle_B(const float* __restrict__ T1, const float* __restrict__ M1,
          _Float16* __restrict__ W)
{
    const int tid  = threadIdx.x;
    const int lane = tid & 31;
    const int w    = blockIdx.x * 8 + (tid >> 5);   // 0 .. NCHUNKS*16-1
    const int c    = w >> 4;                        // chunk
    const int ntg  = w & 15;                        // ntile (16 cols)
    const int k0   = c << 5;

    const float* __restrict__ Brow = (k0 < K1)
        ? (T1 + (size_t)k0 * OUT_DIM)
        : (M1 + (size_t)(k0 - K1) * OUT_DIM);

    const int l15   = lane & 15;
    const int lhalf = lane >> 4;
    const float* __restrict__ bp =
        Brow + (size_t)(lhalf * 16) * OUT_DIM + (ntg * 16 + l15);

    v16h frag;
#pragma unroll
    for (int h = 0; h < 16; ++h)
        frag[h] = (_Float16)bp[(size_t)h * OUT_DIM];

    *(v16h*)&W[((size_t)w * 32 + lane) * 16] = frag;   // coalesced 1KB/wave
}

// --------------------------------------------------------------------------
// GEMM: C[4096,256] += A[4096,33024] @ B, A built on the fly from
// signal (x) feature, B from pre-swizzled f16 (fast) or raw f32 (fallback).
// --------------------------------------------------------------------------
template <bool USE_W>
__global__ void __launch_bounds__(256)
meta_bilinear_gemm(const float* __restrict__ signal,   // [4096,128]
                   const float* __restrict__ feature,  // [4096,256]
                   const float* __restrict__ T1,       // [128, 65536]
                   const float* __restrict__ M1,       // [256, 256]
                   const _Float16* __restrict__ W,     // swizzled B (or null)
                   float* __restrict__ out)            // [4096,256] f32, zeroed
{
    __shared__ __align__(16) _Float16 featLds[MTILE * FEAT_PITCH]; // 34816 B
    __shared__ __align__(16) _Float16 sigLds [MTILE * SIG_PITCH];  // 17408 B

    const int bx      = blockIdx.x;
    const int ks      = bx & (KSPLIT - 1);   // K-slice 0..3
    const int mb      = bx >> 2;             // M-block 0..63
    const int rowBase = mb * MTILE;
    const int tid     = threadIdx.x;

    // ---- stage feature rows (f32 -> f16) into LDS, coalesced float4 ----
    for (int idx = tid; idx < MTILE * (IN_DIM / 4); idx += 256) {
        const int r  = idx / (IN_DIM / 4);
        const int c4 = idx % (IN_DIM / 4);
        const float4 v = ((const float4*)(feature + (size_t)(rowBase + r) * IN_DIM))[c4];
        v4h h = { (_Float16)v.x, (_Float16)v.y, (_Float16)v.z, (_Float16)v.w };
        *(v4h*)&featLds[r * FEAT_PITCH + c4 * 4] = h;
    }
    // ---- stage signal rows ----
    for (int idx = tid; idx < MTILE * (S_DIM / 4); idx += 256) {
        const int r  = idx / (S_DIM / 4);
        const int c4 = idx % (S_DIM / 4);
        const float4 v = ((const float4*)(signal + (size_t)(rowBase + r) * S_DIM))[c4];
        v4h h = { (_Float16)v.x, (_Float16)v.y, (_Float16)v.z, (_Float16)v.w };
        *(v4h*)&sigLds[r * SIG_PITCH + c4 * 4] = h;
    }
    // "ones" column at s = 128 folds the M1 bias into the same GEMM loop
    if (tid < MTILE) sigLds[tid * SIG_PITCH + S_DIM] = (_Float16)1.0f;
    __syncthreads();

    // ---- wave decode: 8 waves = 2(M) x 4(N); wave tile = 32 x 64 ----
    const int wave  = tid >> 5;
    const int lane  = tid & 31;
    const int wm    = wave & 1;        // 0..1  -> M offset 32*wm
    const int wn    = wave >> 1;       // 0..3  -> N offset 64*wn
    const int lhalf = lane >> 4;       // lane group 0/1
    const int l15   = lane & 15;

    const int r0    = wm * 32 + l15;   // LDS row, M-subtile 0
    const int r1    = r0 + 16;         //          M-subtile 1
    const int nCol  = wn * 64 + l15;   // output column base (+ nt*16)

    v8f acc[2][4] = {};                // 8 f32 16x16 accumulators

    const int cStart = ks * CPS;
    for (int c = cStart; c < cStart + CPS; ++c) {
        const int k0 = c << 5;                 // K base of this chunk of 32
        const int s  = k0 >> 8;                // 0..128 (128 => bias region)
        const int i0 = k0 & 255;               // feature column base

        // ---- build A fragments in contiguous k order (same as B) ----
        // lane<16 : halfs 0-15 -> K 0-15
        // lane>=16: halfs 0-15 -> K 16-31
        const int cb = i0 + (lhalf << 4);
        const v8h f0a = *(const v8h*)&featLds[r0 * FEAT_PITCH + cb];
        const v8h f0b = *(const v8h*)&featLds[r0 * FEAT_PITCH + cb + 8];
        const v8h f1a = *(const v8h*)&featLds[r1 * FEAT_PITCH + cb];
        const v8h f1b = *(const v8h*)&featLds[r1 * FEAT_PITCH + cb + 8];
        const _Float16 sg0 = sigLds[r0 * SIG_PITCH + s];
        const _Float16 sg1 = sigLds[r1 * SIG_PITCH + s];

        v16h f0 = __builtin_shufflevector(f0a, f0b, 0,1,2,3,4,5,6,7,
                                                    8,9,10,11,12,13,14,15);
        v16h f1 = __builtin_shufflevector(f1a, f1b, 0,1,2,3,4,5,6,7,
                                                    8,9,10,11,12,13,14,15);
        v16h s0v, s1v;
#pragma unroll
        for (int h = 0; h < 16; ++h) { s0v[h] = sg0; s1v[h] = sg1; }
        const v16h a0 = s0v * f0;      // packed f16 muls (v_pk_mul_f16)
        const v16h a1 = s1v * f1;

        // ---- 4 B fragments + 8 WMMAs ----
#pragma unroll
        for (int nt = 0; nt < 4; ++nt) {
            v16h b;
            if (USE_W) {
                // fragment is 32B contiguous, coalesced across the wave
                const size_t off = ((size_t)(c * 16 + wn * 4 + nt) * 32 + lane) * 16;
                b = *(const v16h*)&W[off];
            } else {
                const float* __restrict__ Brow = (k0 < K1)
                    ? (T1 + (size_t)k0 * OUT_DIM)
                    : (M1 + (size_t)(k0 - K1) * OUT_DIM);
                const float* __restrict__ bp =
                    Brow + (size_t)(lhalf * 16) * OUT_DIM + (nCol + nt * 16);
#pragma unroll
                for (int h = 0; h < 16; ++h)
                    b[h] = (_Float16)bp[(size_t)h * OUT_DIM];
            }
            acc[0][nt] = __builtin_amdgcn_wmma_f32_16x16x32_f16(
                false, a0, false, b, (short)0, acc[0][nt], false, false);
            acc[1][nt] = __builtin_amdgcn_wmma_f32_16x16x32_f16(
                false, a1, false, b, (short)0, acc[1][nt], false, false);
        }
    }

    // ---- epilogue: C/D layout (VGPR r, lanes 0-15: M=r; lanes 16-31: M=8+r)
#pragma unroll
    for (int msub = 0; msub < 2; ++msub) {
        const int mg = rowBase + wm * 32 + msub * 16 + lhalf * 8;
#pragma unroll
        for (int nt = 0; nt < 4; ++nt) {
            const int n = wn * 64 + nt * 16 + l15;
#pragma unroll
            for (int r = 0; r < 8; ++r)
                atomicAdd(&out[(size_t)(mg + r) * OUT_DIM + n], acc[msub][nt][r]);
        }
    }
}

extern "C" void kernel_launch(void* const* d_in, const int* in_sizes, int n_in,
                              void* d_out, int out_size, void* d_ws, size_t ws_size,
                              hipStream_t stream) {
    (void)in_sizes; (void)n_in; (void)out_size;
    const float* signal  = (const float*)d_in[0];
    const float* feature = (const float*)d_in[1];
    const float* T1      = (const float*)d_in[2];
    const float* M1      = (const float*)d_in[3];
    float* out = (float*)d_out;

    // zero output (K-split slices accumulate with atomics); capturable memset
    hipMemsetAsync(out, 0, (size_t)BATCH * OUT_DIM * sizeof(float), stream);

    dim3 block(256);
    if (ws_size >= W_BYTES) {
        _Float16* W = (_Float16*)d_ws;
        // NCHUNKS*16 wave-tasks, 8 waves per block
        dim3 gswz((NCHUNKS * 16) / 8);                  // 2064
        hipLaunchKernelGGL(swizzle_B, gswz, block, 0, stream, T1, M1, W);
        dim3 grid(MBLOCKS * KSPLIT);                    // 256
        hipLaunchKernelGGL((meta_bilinear_gemm<true>), grid, block, 0, stream,
                           signal, feature, T1, M1, W, out);
    } else {
        dim3 grid(MBLOCKS * KSPLIT);
        hipLaunchKernelGGL((meta_bilinear_gemm<false>), grid, block, 0, stream,
                           signal, feature, T1, M1, (const _Float16*)nullptr, out);
    }
}
